// SwinTransformerBlock_20804821582178
// MI455X (gfx1250) — matrix operand order repaired and checked by reading, not compiled, not run on hardware
//
#include <hip/hip_runtime.h>
#include <hip/hip_bf16.h>
#include <math.h>

// ---------------------------------------------------------------------------
// CDNA5 (gfx1250) fused Swin block: wave32, WMMA bf16 16x16x32, 8 waves/block.
// One workgroup == one 8x8 window (64 tokens x 256 channels).
// Weights are pre-converted fp32 -> bf16 into d_ws by a prep kernel so the
// GEMM hot loops issue only 128-bit loads + v_wmma. All fp32->bf16 paths use
// native fptrunc (vector convertvector) so the backend can select the
// hardware bf16 converters.
// ---------------------------------------------------------------------------

typedef __attribute__((ext_vector_type(16))) __bf16 bf16x16;
typedef __attribute__((ext_vector_type(8)))  __bf16 bf16x8;
typedef __attribute__((ext_vector_type(2)))  __bf16 bf16x2;
typedef __attribute__((ext_vector_type(8)))  float  f32x8;
typedef __attribute__((ext_vector_type(2)))  float  f32x2;

union BFrag  { bf16x16 v; unsigned short u[16]; uint4 q[2]; };
union Pack8  { uint4 q; bf16x8 b; unsigned short s[8]; };
union Pack2  { unsigned int u; bf16x2 b; };

// fp32x8 -> bf16x8 via native fptrunc (RNE).
__device__ __forceinline__ Pack8 cvt8(f32x8 v) {
    Pack8 pk;
    pk.b = __builtin_convertvector(v, bf16x8);
    return pk;
}

#define WMMA_BF16(a, b, c)                                                   \
    __builtin_amdgcn_wmma_f32_16x16x32_bf16(false, (a), false, (b),          \
                                            (short)0, (c), false, false)

// A fragment: 16x32 bf16 tile, row-major source (stride lda elements).
// ISA layout: lane {hf=lane>>4, m=lane&15}; elements 0..7 = K 8hf..8hf+7,
// elements 8..15 = K 16+8hf..16+8hf+7  -> two contiguous 16B runs.
__device__ __forceinline__ bf16x16 load_a(const unsigned short* A, int lda,
                                          int m0, int k0) {
    const int lane = threadIdx.x & 31;
    const unsigned short* p = A + (m0 + (lane & 15)) * lda + k0 + 8 * (lane >> 4);
    BFrag f;
    f.q[0] = *(const uint4*)(p);
    f.q[1] = *(const uint4*)(p + 16);
    return f.v;
}

// B fragment (32x16, K x N) from row-major [N, K] source (i.e. B = M^T).
// ISA layout: n = lane&15, elements 0..7 = K 8hf..8hf+7,
// elements 8..15 = K 16+8hf..16+8hf+7  -> two contiguous 16B runs.
__device__ __forceinline__ bf16x16 load_bT(const unsigned short* M, int ldm,
                                           int n0, int k0) {
    const int lane = threadIdx.x & 31;
    const unsigned short* p = M + (size_t)(n0 + (lane & 15)) * ldm + k0 + 8 * (lane >> 4);
    BFrag f;
    f.q[0] = *(const uint4*)(p);
    f.q[1] = *(const uint4*)(p + 16);
    return f.v;
}

// Per-token layernorm over 256 channels -> bf16 (packed 128-bit stores).
__device__ __forceinline__ void layernorm64x256(const float* resid,
                                                unsigned short* dst,
                                                const float* w, const float* bias,
                                                int tid) {
    if (tid < 64) {
        const float* r = resid + tid * 256;
        float m = 0.f;
        for (int c = 0; c < 256; ++c) m += r[c];
        m *= (1.f / 256.f);
        float v = 0.f;
        for (int c = 0; c < 256; ++c) { float d = r[c] - m; v += d * d; }
        v *= (1.f / 256.f);
        float rs = rsqrtf(v + 1e-5f);
        unsigned short* h = dst + tid * 256;
        for (int c = 0; c < 256; c += 8) {
            f32x8 v8;
#pragma unroll
            for (int j = 0; j < 8; ++j)
                v8[j] = (r[c + j] - m) * rs * w[c + j] + bias[c + j];
            *(uint4*)(h + c) = cvt8(v8).q;
        }
    }
}

// ---------------- prep kernel: fp32 weights -> bf16 in workspace ----------
#define IPW_N  (768 * 256)
#define OPW_N  (256 * 256)
#define W1_N   (1024 * 256)
#define W2_N   (256 * 1024)
#define CVT_TOTAL (IPW_N + OPW_N + W1_N + W2_N)     // 786432 elements

__global__ __launch_bounds__(256)
void cvt_weights_bf16(const float* __restrict__ ipw, const float* __restrict__ opw,
                      const float* __restrict__ w1,  const float* __restrict__ w2,
                      unsigned int* __restrict__ ws) {          // packed bf16x2
    int i = (blockIdx.x * 256 + threadIdx.x) * 2;               // element index
    if (i >= CVT_TOTAL) return;
    const float* src; int off;
    if      (i < IPW_N)                { src = ipw; off = i; }
    else if (i < IPW_N + OPW_N)        { src = opw; off = i - IPW_N; }
    else if (i < IPW_N + OPW_N + W1_N) { src = w1;  off = i - IPW_N - OPW_N; }
    else                               { src = w2;  off = i - IPW_N - OPW_N - W1_N; }
    f32x2 v; v[0] = src[off]; v[1] = src[off + 1];
    Pack2 cv; cv.b = __builtin_convertvector(v, bf16x2);
    ws[i >> 1] = cv.u;
}

// ---------------- main fused kernel ---------------------------------------
__global__ __launch_bounds__(256, 1)
void swin_block_gfx1250(const float* __restrict__ x,
                        const float* __restrict__ ln1w, const float* __restrict__ ln1b,
                        const unsigned short* __restrict__ ipw,  // bf16 [768,256]
                        const float* __restrict__ ipb,
                        const unsigned short* __restrict__ opw,  // bf16 [256,256]
                        const float* __restrict__ opb,
                        const float* __restrict__ ln2w, const float* __restrict__ ln2b,
                        const unsigned short* __restrict__ w1,   // bf16 [1024,256]
                        const float* __restrict__ b1,
                        const unsigned short* __restrict__ w2,   // bf16 [256,1024]
                        const float* __restrict__ b2,
                        float* __restrict__ out)
{
    extern __shared__ char smem[];
    float*          resid = (float*)smem;                     // 64*256 f32   (64 KB)
    unsigned short* hbuf  = (unsigned short*)(smem + 65536);  // 64*256 bf16  (32 KB)
    char*           big   = smem + 98304;                     // 128 KB region
    unsigned short* qk    = (unsigned short*)big;             // 64*512 bf16: Q|K (64 KB)
    unsigned short* vT    = (unsigned short*)(big + 65536);   // 256*64 bf16: V^T (32 KB)
    float*          sc    = (float*)(big + 98304);            // 64*64 f32    (16 KB)
    unsigned short* pbf   = (unsigned short*)(big + 114688);  // 64*64 bf16   ( 8 KB)
    unsigned short* mid   = (unsigned short*)big;             // 64*1024 bf16 (reuses big)

    const int tid  = threadIdx.x;
    const int wave = tid >> 5;
    const int lane = tid & 31;
    const int ln15 = lane & 15;
    const int hf   = lane >> 4;

    const int widx = blockIdx.x;          // 2048 windows
    const int bb_  = widx >> 8;           // batch
    const int wh   = (widx >> 4) & 15;
    const int ww   = widx & 15;

    // Seed L2 with bf16 weights (lowers to global_prefetch_b8).
    __builtin_prefetch(ipw + tid * 256, 0, 1);
    __builtin_prefetch(w1  + tid * 256, 0, 1);

    // ---- stage 0: gather cyclically-shifted window, NCHW -> [t][c] fp32 ----
    for (int i = tid; i < 64 * 256; i += 256) {
        int c = i >> 6, t = i & 63;
        int h = ((wh << 3) + (t >> 3) + 4) & 127;   // roll(-4)
        int w = ((ww << 3) + (t & 7) + 4) & 127;
        resid[t * 256 + c] = x[(((size_t)bb_ * 256 + c) << 14) + (h << 7) + w];
    }
    __syncthreads();

    // ---- stage 1: LN1 -> hbuf (bf16) ----
    layernorm64x256(resid, hbuf, ln1w, ln1b, tid);
    __syncthreads();

    // ---- stage 2: QKV = LN1 @ Wqkv^T + b.  Q,K -> qk[64][512], V -> vT ----
    for (int tile = wave; tile < 192; tile += 8) {      // 4 M-tiles x 48 N-tiles
        int mi = tile & 3, ni = tile >> 2;
        f32x8 acc;
        float bv = ipb[ni * 16 + ln15];
#pragma unroll
        for (int r = 0; r < 8; ++r) acc[r] = bv;
#pragma unroll
        for (int k0 = 0; k0 < 256; k0 += 32) {
            bf16x16 a  = load_a(hbuf, 256, mi * 16, k0);
            bf16x16 bt = load_bT(ipw, 256, ni * 16, k0);
            acc = WMMA_BF16(a, bt, acc);
        }
        Pack8 pk = cvt8(acc);
        if (ni < 32) {                                   // Q or K: [t][c] layout
            unsigned short* d = qk + (mi * 16 + 8 * hf) * 512 + ni * 16 + ln15;
#pragma unroll
            for (int r = 0; r < 8; ++r) d[r * 512] = pk.s[r];
        } else {                                         // V: transposed [c][t]
            int cv = (ni - 32) * 16 + ln15;              // 0..255
            *(uint4*)(vT + cv * 64 + mi * 16 + 8 * hf) = pk.q;
        }
    }
    __syncthreads();

    // ---- stage 3: attention, one head at a time (8 heads, dh=32) ----
    const float scale = 0.17677669529663687f;           // 1/sqrt(32)
    for (int hd = 0; hd < 8; ++hd) {
        // S = Q K^T * scale : 16 tiles, K-dim 32 -> single WMMA each
        for (int tile = wave; tile < 16; tile += 8) {
            int mi = tile >> 2, ni = tile & 3;
            f32x8 acc = {};
            bf16x16 a  = load_a(qk, 512, mi * 16, hd * 32);            // Q
            bf16x16 bt = load_bT(qk + 256, 512, ni * 16, hd * 32);     // K^T
            acc = WMMA_BF16(a, bt, acc);
            float* d = sc + (mi * 16 + 8 * hf) * 64 + ni * 16 + ln15;
#pragma unroll
            for (int r = 0; r < 8; ++r) d[r * 64] = acc[r] * scale;
        }
        __syncthreads();

        // row softmax (64 rows) -> pbf bf16
        if (tid < 64) {
            float* row = sc + tid * 64;
            float mx = row[0];
            for (int j = 1; j < 64; ++j) mx = fmaxf(mx, row[j]);
            float s = 0.f;
            for (int j = 0; j < 64; ++j) { float e = __expf(row[j] - mx); row[j] = e; s += e; }
            float inv = 1.f / s;
            unsigned short* pr = pbf + tid * 64;
            for (int j = 0; j < 64; j += 8) {
                f32x8 v8;
#pragma unroll
                for (int u = 0; u < 8; ++u) v8[u] = row[j + u] * inv;
                *(uint4*)(pr + j) = cvt8(v8).q;
            }
        }
        __syncthreads();

        // O = P @ V : 8 tiles (4 x 2), one per wave, K-dim 64
        {
            int mi = wave >> 1, ni = wave & 1;
            f32x8 acc = {};
#pragma unroll
            for (int k0 = 0; k0 < 64; k0 += 32) {
                bf16x16 a  = load_a(pbf, 64, mi * 16, k0);
                bf16x16 bv = load_bT(vT + hd * 32 * 64, 64, ni * 16, k0); // V^T rows
                acc = WMMA_BF16(a, bv, acc);
            }
            Pack8 pk = cvt8(acc);
            unsigned short* d = hbuf + (mi * 16 + 8 * hf) * 256 + hd * 32 + ni * 16 + ln15;
#pragma unroll
            for (int r = 0; r < 8; ++r) d[r * 256] = pk.s[r];
        }
        __syncthreads();
    }

    // ---- stage 4: out-proj + residual into resid ----
    for (int tile = wave; tile < 64; tile += 8) {
        int mi = tile & 3, ni = tile >> 2;
        f32x8 acc;
        float bv = opb[ni * 16 + ln15];
#pragma unroll
        for (int r = 0; r < 8; ++r) acc[r] = bv;
#pragma unroll
        for (int k0 = 0; k0 < 256; k0 += 32) {
            bf16x16 a  = load_a(hbuf, 256, mi * 16, k0);
            bf16x16 bt = load_bT(opw, 256, ni * 16, k0);
            acc = WMMA_BF16(a, bt, acc);
        }
        float* d = resid + (mi * 16 + 8 * hf) * 256 + ni * 16 + ln15;
#pragma unroll
        for (int r = 0; r < 8; ++r) d[r * 256] += acc[r];
    }
    __syncthreads();

    // ---- stage 5: LN2 -> hbuf ----
    layernorm64x256(resid, hbuf, ln2w, ln2b, tid);
    __syncthreads();

    // ---- stage 6: MLP fc1 (256 -> 1024) + exact GELU -> mid bf16 ----
    for (int tile = wave; tile < 256; tile += 8) {      // 4 x 64 tiles
        int mi = tile & 3, ni = tile >> 2;
        f32x8 acc;
        float bv = b1[ni * 16 + ln15];
#pragma unroll
        for (int r = 0; r < 8; ++r) acc[r] = bv;
#pragma unroll
        for (int k0 = 0; k0 < 256; k0 += 32) {
            bf16x16 a  = load_a(hbuf, 256, mi * 16, k0);
            bf16x16 bt = load_bT(w1, 256, ni * 16, k0);
            acc = WMMA_BF16(a, bt, acc);
        }
        f32x8 g;
#pragma unroll
        for (int r = 0; r < 8; ++r)
            g[r] = 0.5f * acc[r] * (1.f + erff(acc[r] * 0.70710678118654752f));
        Pack8 pk = cvt8(g);
        unsigned short* d = mid + (mi * 16 + 8 * hf) * 1024 + ni * 16 + ln15;
#pragma unroll
        for (int r = 0; r < 8; ++r) d[r * 1024] = pk.s[r];
    }
    __syncthreads();

    // ---- stage 7: MLP fc2 (1024 -> 256) + residual ----
    for (int tile = wave; tile < 64; tile += 8) {
        int mi = tile & 3, ni = tile >> 2;
        f32x8 acc;
        float bv = b2[ni * 16 + ln15];
#pragma unroll
        for (int r = 0; r < 8; ++r) acc[r] = bv;
#pragma unroll 4
        for (int k0 = 0; k0 < 1024; k0 += 32) {
            bf16x16 a  = load_a(mid, 1024, mi * 16, k0);
            bf16x16 bt = load_bT(w2, 1024, ni * 16, k0);
            acc = WMMA_BF16(a, bt, acc);
        }
        float* d = resid + (mi * 16 + 8 * hf) * 256 + ni * 16 + ln15;
#pragma unroll
        for (int r = 0; r < 8; ++r) d[r * 256] += acc[r];
    }
    __syncthreads();

    // ---- stage 8: scatter back (inverse shift == same source coords) ----
    for (int i = tid; i < 64 * 256; i += 256) {
        int c = i >> 6, t = i & 63;
        int h = ((wh << 3) + (t >> 3) + 4) & 127;
        int w = ((ww << 3) + (t & 7) + 4) & 127;
        out[(((size_t)bb_ * 256 + c) << 14) + (h << 7) + w] = resid[t * 256 + c];
    }
}

extern "C" void kernel_launch(void* const* d_in, const int* in_sizes, int n_in,
                              void* d_out, int out_size, void* d_ws, size_t ws_size,
                              hipStream_t stream) {
    (void)in_sizes; (void)n_in; (void)out_size; (void)ws_size;
    const float* x    = (const float*)d_in[0];
    const float* ln1w = (const float*)d_in[1];
    const float* ln1b = (const float*)d_in[2];
    const float* ipw  = (const float*)d_in[3];
    const float* ipb  = (const float*)d_in[4];
    const float* opw  = (const float*)d_in[5];
    const float* opb  = (const float*)d_in[6];
    const float* ln2w = (const float*)d_in[7];
    const float* ln2b = (const float*)d_in[8];
    const float* w1   = (const float*)d_in[9];
    const float* b1   = (const float*)d_in[10];
    const float* w2   = (const float*)d_in[11];
    const float* b2   = (const float*)d_in[12];
    float* out = (float*)d_out;

    unsigned short* ws = (unsigned short*)d_ws;     // bf16 weights, 1.5 MB
    unsigned short* ipw_bf = ws;
    unsigned short* opw_bf = ws + IPW_N;
    unsigned short* w1_bf  = ws + IPW_N + OPW_N;
    unsigned short* w2_bf  = ws + IPW_N + OPW_N + W1_N;

    // 1) convert fp32 weights -> bf16 (once per launch, stream-ordered)
    cvt_weights_bf16<<<(CVT_TOTAL / 2 + 255) / 256, 256, 0, stream>>>(
        ipw, opw, w1, w2, (unsigned int*)ws);

    // 2) fused Swin block
    const size_t shmem = 229376;   // 64KB resid + 32KB hbuf + 128KB big
    dim3 grid(2048), block(256);
    swin_block_gfx1250<<<grid, block, shmem, stream>>>(
        x, ln1w, ln1b, ipw_bf, ipb, opw_bf, opb, ln2w, ln2b,
        w1_bf, b1, w2_bf, b2, out);
}
